// HopfieldMemory_76089640616055
// MI455X (gfx1250) — matrix operand order repaired and checked by reading, not compiled, not run on hardware
//
#include <hip/hip_runtime.h>
#include <hip/hip_fp16.h>

typedef __attribute__((ext_vector_type(16))) _Float16 v16h;
typedef __attribute__((ext_vector_type(8)))  float    v8f;
typedef __attribute__((ext_vector_type(4)))  int      v4i;

#define DIM 128          // feature dim (fixed by reference)
#define M_WG 64          // queries per workgroup (8 waves: 4 M-subtiles x 2 N-subtiles)
#define NCHUNKS 16       // key-dimension split for parallelism
#define BETA 8.0f

#ifndef __has_builtin
#define __has_builtin(x) 0
#endif

#if __has_builtin(__builtin_amdgcn_global_load_async_to_lds_b128)
#define USE_ASYNC 1
// pointer types matching the builtin's parameters: v4i in AS1 (global) / AS3 (LDS)
typedef __attribute__((address_space(1))) v4i gv4i;
typedef __attribute__((address_space(3))) v4i lv4i;
#else
#define USE_ASYNC 0
#endif

__device__ __forceinline__ void wait_async0() {
#if __has_builtin(__builtin_amdgcn_s_wait_asynccnt)
    __builtin_amdgcn_s_wait_asynccnt(0);
#elif USE_ASYNC
    asm volatile("s_wait_asynccnt 0x0" ::: "memory");
#endif
}

__device__ __forceinline__ v8f wmma_f16(v16h a, v16h b, v8f c) {
    return __builtin_amdgcn_wmma_f32_16x16x32_f16(false, a, false, b,
                                                  (short)0, c, false, false);
}

// ---------------------------------------------------------------------------
// Zero a float buffer
// ---------------------------------------------------------------------------
__global__ void hop_zero_f32(float* __restrict__ p, int n) {
    int i = blockIdx.x * blockDim.x + threadIdx.x;
    if (i < n) p[i] = 0.0f;
}

// ---------------------------------------------------------------------------
// Pre-pass: keys f32 -> f16 (halves hot-loop key traffic) and rk = 1/||k||.
// 32 rows per 256-thread block; 8 threads x 16 elems per row.
// ---------------------------------------------------------------------------
__global__ void __launch_bounds__(256)
hop_prep_keys(const float* __restrict__ keys, _Float16* __restrict__ keys16,
              float* __restrict__ rk, int N)
{
    __shared__ float ps[256];
    const int t = threadIdx.x;
    const int row = blockIdx.x * 32 + (t >> 3);
    const int cb = t & 7;
    float ss = 0.0f;
    if (row < N) {
        const float* src = keys + (size_t)row * DIM + cb * 16;
        _Float16*    dst = keys16 + (size_t)row * DIM + cb * 16;
        #pragma unroll
        for (int j = 0; j < 16; ++j) {
            float v = src[j]; ss += v * v; dst[j] = (_Float16)v;
        }
    }
    ps[t] = ss;
    __syncthreads();
    if (t < 32) {
        int r = blockIdx.x * 32 + t;
        if (r < N) {
            float s = 0.0f;
            #pragma unroll
            for (int j = 0; j < 8; ++j) s += ps[t * 8 + j];
            rk[r] = rsqrtf(fmaxf(s, 1e-24f));   // == 1/max(||k||, 1e-12)
        }
    }
}

// ---------------------------------------------------------------------------
// Main fused kernel. Grid: (NCHUNKS, B/M_WG). Block: 256 threads = 8 waves.
// LDS: numer[M_WG*C] | ldsK[2][32*DIM] f16 (double buffer) | psum[256] | rq | denom
// ---------------------------------------------------------------------------
__global__ void __launch_bounds__(256)
hopfield_main(const float* __restrict__ emb, const _Float16* __restrict__ keys16,
              const float* __restrict__ rk, const int* __restrict__ labels,
              float* __restrict__ denomOut, float* __restrict__ partials,
              int B, int N, int C, int chunkLen)
{
    extern __shared__ char smem[];
    float*    numer  = (float*)smem;                       // M_WG*C floats
    _Float16* ldsK   = (_Float16*)(numer + M_WG * C);      // 2 * 32*DIM halves
    float*    psum   = (float*)(ldsK + 2 * 32 * DIM);      // 256 floats
    float*    rqLds  = psum + 256;                         // M_WG floats
    float*    denomL = rqLds + M_WG;                       // M_WG floats

    const int t    = threadIdx.x;
    const int lane = t & 31;
    const int wave = t >> 5;
    const int msub = wave & 3;
    const int nsub = wave >> 2;
    const int hi   = (lane >> 4);
    const int qBase = blockIdx.y * M_WG;
    const int chunk = blockIdx.x;

    for (int i = t; i < M_WG * C; i += 256) numer[i] = 0.0f;
    if (t < M_WG) denomL[t] = 0.0f;

    // ---- query norms (one-time): 4 threads x 32 elems per row ----
    {
        int row = t >> 2, qtr = t & 3;
        const float* src = emb + (size_t)(qBase + row) * DIM + qtr * 32;
        float ss = 0.0f;
        #pragma unroll
        for (int j = 0; j < 32; ++j) { float v = src[j]; ss += v * v; }
        psum[t] = ss;
    }
    __syncthreads();
    if (t < M_WG) {
        float s = psum[t*4] + psum[t*4+1] + psum[t*4+2] + psum[t*4+3];
        rqLds[t] = rsqrtf(fmaxf(s, 1e-24f));
    }
    __syncthreads();

    // ---- A fragments (16x32 f16, contiguous k = 16*hi + i, same as B), loaded once ----
    v16h a[4];
    {
        const int row  = qBase + msub * 16 + (lane & 15);
        const int koff = hi * 16;
        const float* qrow = emb + (size_t)row * DIM;
        #pragma unroll
        for (int kb = 0; kb < 4; ++kb) {
            v16h av;
            #pragma unroll
            for (int i = 0; i < 16; ++i) {
                int K = kb * 32 + koff + i;
                av[i] = (_Float16)qrow[K];
            }
            a[kb] = av;
        }
    }
    float rqv[8];
    #pragma unroll
    for (int r = 0; r < 8; ++r) rqv[r] = rqLds[msub * 16 + hi * 8 + r];
    float dacc[8] = {0,0,0,0,0,0,0,0};

    const int chunkStart = chunk * chunkLen;
    const int chunkEnd   = min(chunkStart + chunkLen, N);
    const int nTiles     = (chunkEnd > chunkStart) ? ((chunkEnd - chunkStart + 31) >> 5) : 0;

    const uint4 zero4 = {0u, 0u, 0u, 0u};

    // ---- compute one 64x32 tile from LDS buffer ----
    auto computeTile = [&](const _Float16* kbuf, int base) {
        const int keyLocal = nsub * 16 + (lane & 15);
        const int n = base + keyLocal;
        const bool valid = (n < chunkEnd);
        const float rkv = valid ? rk[n] : 0.0f;
        const int   lab = valid ? labels[n] : 0;

        const _Float16* bsrc = kbuf + keyLocal * DIM + hi * 16;
        v16h b0 = *(const v16h*)(bsrc);
        v16h b1 = *(const v16h*)(bsrc + 32);
        v16h b2 = *(const v16h*)(bsrc + 64);
        v16h b3 = *(const v16h*)(bsrc + 96);
        v8f c = {};
        c = wmma_f16(a[0], b0, c);
        c = wmma_f16(a[1], b1, c);
        c = wmma_f16(a[2], b2, c);
        c = wmma_f16(a[3], b3, c);

        #pragma unroll
        for (int r = 0; r < 8; ++r) {
            float w = __expf(BETA * c[r] * rqv[r] * rkv);
            w = valid ? w : 0.0f;
            dacc[r] += w;
            atomicAdd(&numer[(msub * 16 + hi * 8 + r) * C + lab], w);  // ds_add_f32
        }
    };

#if USE_ASYNC
    // ---- async double-buffered pipeline (GLOBAL_LOAD_ASYNC_TO_LDS_B128) ----
    auto issueTile = [&](int it, int bufIdx) {
        const int base = chunkStart + it * 32;
        const int row  = base + (t >> 3);
        const _Float16* g = keys16 + (size_t)base * DIM + t * 16;
        _Float16*       l = ldsK + bufIdx * (32 * DIM) + t * 16;
        if (row < chunkEnd) {
            __builtin_amdgcn_global_load_async_to_lds_b128(
                (gv4i*)g,       (lv4i*)l,       0, 0);
            __builtin_amdgcn_global_load_async_to_lds_b128(
                (gv4i*)(g + 8), (lv4i*)(l + 8), 0, 0);
        } else {
            ((uint4*)l)[0] = zero4;
            ((uint4*)l)[1] = zero4;
        }
    };
    if (nTiles > 0) {
        issueTile(0, 0);
        wait_async0();
        __syncthreads();
        for (int it = 0; it < nTiles; ++it) {
            const int cur = it & 1;
            if (it + 1 < nTiles) issueTile(it + 1, cur ^ 1);
            computeTile(ldsK + cur * (32 * DIM), chunkStart + it * 32);
            wait_async0();
            __syncthreads();
        }
    }
#else
    // ---- register-staged double-buffered pipeline (sync loads) ----
    auto loadRegs = [&](int it, uint4* r01) {
        const int base = chunkStart + it * 32;
        const int row  = base + (t >> 3);
        const uint4* g = (const uint4*)(keys16 + (size_t)base * DIM + t * 16);
        r01[0] = (row < chunkEnd) ? g[0] : zero4;
        r01[1] = (row < chunkEnd) ? g[1] : zero4;
    };
    if (nTiles > 0) {
        uint4 stage[2];
        loadRegs(0, stage);
        {
            uint4* l = (uint4*)(ldsK + t * 16);
            l[0] = stage[0]; l[1] = stage[1];
        }
        __syncthreads();
        for (int it = 0; it < nTiles; ++it) {
            const int cur = it & 1;
            const bool more = (it + 1 < nTiles);
            if (more) loadRegs(it + 1, stage);
            computeTile(ldsK + cur * (32 * DIM), chunkStart + it * 32);
            if (more) {
                uint4* l = (uint4*)(ldsK + (cur ^ 1) * (32 * DIM) + t * 16);
                l[0] = stage[0]; l[1] = stage[1];
            }
            __syncthreads();
        }
    }
#endif

    // ---- denominator reduction ----
    #pragma unroll
    for (int r = 0; r < 8; ++r)
        atomicAdd(&denomL[msub * 16 + hi * 8 + r], dacc[r]);
    __syncthreads();
    if (t < M_WG) atomicAdd(&denomOut[qBase + t], denomL[t]);

    // ---- flush class bins to per-chunk partials (coalesced stores) ----
    float* pdst = partials + ((size_t)chunk * B + qBase) * C;
    for (int i = t; i < M_WG * C; i += 256) pdst[i] = numer[i];
}

// ---------------------------------------------------------------------------
// Reduce NCHUNKS partials and divide by softmax denominator
// ---------------------------------------------------------------------------
__global__ void hopfield_finalize(const float* __restrict__ partials,
                                  const float* __restrict__ denom,
                                  float* __restrict__ out, int B, int C, int nchunks)
{
    int idx = blockIdx.x * blockDim.x + threadIdx.x;
    if (idx >= B * C) return;
    size_t stride = (size_t)B * C;
    float s = 0.0f;
    for (int ch = 0; ch < nchunks; ++ch) s += partials[ch * stride + idx];
    out[idx] = s / denom[idx / C];
}

// ---------------------------------------------------------------------------
extern "C" void kernel_launch(void* const* d_in, const int* in_sizes, int n_in,
                              void* d_out, int out_size, void* d_ws, size_t ws_size,
                              hipStream_t stream) {
    const float* emb    = (const float*)d_in[0];
    const float* keys   = (const float*)d_in[1];
    const int*   labels = (const int*)d_in[2];
    float* out = (float*)d_out;
    float* ws  = (float*)d_ws;

    const int B = in_sizes[0] / DIM;     // 1024
    const int N = in_sizes[1] / DIM;     // 100000
    const int C = out_size / B;          // 1000

    // workspace layout (float units, 256-float aligned sections)
    auto align256 = [](size_t x) { return (x + 255) & ~(size_t)255; };
    float*    denom    = ws;                                  // B
    size_t    off      = align256((size_t)B);
    float*    rk       = ws + off;                            // N
    off                = align256(off + (size_t)N);
    _Float16* keys16   = (_Float16*)(ws + off);               // N*DIM halves
    off                = align256(off + (size_t)N * DIM / 2);
    float*    partials = ws + off;                            // NCHUNKS*B*C

    const int chunkLen = (N + NCHUNKS - 1) / NCHUNKS;
    const size_t smemBytes = (size_t)M_WG * C * 4   // numer bins
                           + 2 * 32 * DIM * 2       // double-buffered key tile f16
                           + 256 * 4                // psum
                           + M_WG * 4               // rq
                           + M_WG * 4;              // denom
    (void)hipFuncSetAttribute((const void*)hopfield_main,
                              hipFuncAttributeMaxDynamicSharedMemorySize,
                              (int)smemBytes);

    hop_zero_f32<<<(B + 255) / 256, 256, 0, stream>>>(denom, B);
    hop_prep_keys<<<(N + 31) / 32, 256, 0, stream>>>(keys, keys16, rk, N);

    dim3 grid(NCHUNKS, B / M_WG);
    hopfield_main<<<grid, 256, smemBytes, stream>>>(
        emb, keys16, rk, labels, denom, partials, B, N, C, chunkLen);

    int total = B * C;
    hopfield_finalize<<<(total + 255) / 256, 256, 0, stream>>>(
        partials, denom, out, B, C, NCHUNKS);
}
